// MVPFusion_12403865551055
// MI455X (gfx1250) — matrix operand order repaired and checked by reading, not compiled, not run on hardware
//
#include <hip/hip_runtime.h>
#include <hip/hip_bf16.h>

typedef _Float16 h8   __attribute__((ext_vector_type(8)));
typedef _Float16 v16h __attribute__((ext_vector_type(16)));
typedef float    v8f  __attribute__((ext_vector_type(8)));

union H16 { v16h v; h8 h[2]; };

static __device__ __forceinline__ v8f wmma_f16(v16h a, v16h b, v8f c) {
  return __builtin_amdgcn_wmma_f32_16x16x32_f16(false, a, false, b, (short)0, c, false, false);
}

#define CCH 128
#define HW  4096
#define SS  336
#define SP  352   // S padded to multiple of 32

// ---------------- small prep kernels ----------------

__global__ void k_f32_to_f16(const float* __restrict__ in, _Float16* __restrict__ out, int n) {
  int i = blockIdx.x * 256 + threadIdx.x;
  if (i < n) out[i] = (_Float16)in[i];
}

__global__ void k_zero16(_Float16* __restrict__ p, int n) {
  int i = blockIdx.x * 256 + threadIdx.x;
  if (i < n) p[i] = (_Float16)0.f;
}

// tam[s] = sigmoid(sum_c sal_w[c]*glb[c,s] + sal_b)
__global__ void k_tam(const float* __restrict__ xg, const float* __restrict__ sal_w,
                      const float* __restrict__ sal_b, float* __restrict__ tam) {
  int s = blockIdx.x * 256 + threadIdx.x;
  if (s >= HW) return;
  float acc = 0.f;
  #pragma unroll 4
  for (int c = 0; c < CCH; ++c) acc += sal_w[c] * xg[c * HW + s];
  tam[s] = 1.f / (1.f + __expf(-(acc + sal_b[0])));
}

// q[p][l][c] = x[p,c,y,z] * tam[a*32+y/2, b*32+z/2]  (fp32 + fp16 copies)
__global__ void k_build_q(const float* __restrict__ x, const float* __restrict__ tam,
                          _Float16* __restrict__ q16, float* __restrict__ qf32) {
  long idx = (long)blockIdx.x * 256 + threadIdx.x;   // 4*128*4096
  int p = (int)(idx >> 19);
  int c = (int)((idx >> 12) & 127);
  int l = (int)(idx & 4095);
  int y = l >> 6, z = l & 63;
  int a = p >> 1, b = p & 1;
  float t = tam[((a << 5) + (y >> 1)) * 64 + (b << 5) + (z >> 1)];
  float v = x[idx] * t;
  long qi = ((long)p * HW + l) * CCH + c;
  qf32[qi] = v;
  q16[qi] = (_Float16)v;
}

// pooled KV tokens from global patch p: kv16[p][s][c]
__global__ void k_pool(const float* __restrict__ xg, _Float16* __restrict__ kv16) {
  int idx = blockIdx.x * 256 + threadIdx.x;          // 4*336*128
  if (idx >= 4 * SS * CCH) return;
  int c = idx & 127;
  int s = (idx >> 7) % SS;
  int p = idx / (SS * CCH);
  int ksz, tw, t;
  if (s < 256)      { ksz = 2; tw = 16; t = s; }
  else if (s < 320) { ksz = 4; tw = 8;  t = s - 256; }
  else              { ksz = 8; tw = 4;  t = s - 320; }
  int ti = t / tw, tj = t % tw;
  int a = p >> 1, b = p & 1;
  int r0 = (a << 5) + ti * ksz, c0 = (b << 5) + tj * ksz;
  float sum = 0.f;
  for (int dy = 0; dy < ksz; ++dy)
    for (int dz = 0; dz < ksz; ++dz)
      sum += xg[c * HW + (r0 + dy) * 64 + (c0 + dz)];
  kv16[idx] = (_Float16)(sum / (float)(ksz * ksz));
}

// ---------------- small WMMA GEMM (one wave = one 16x16 tile) ----------------
// used for the tiny K/V projections (M=336)
__global__ void k_gemm_h16(const _Float16* __restrict__ A, long sA, int lda,
                           const _Float16* __restrict__ W, long sW,
                           const float* __restrict__ bias, int sB,
                           _Float16* __restrict__ O, long sO, int ldo,
                           int K, int relu, int transposed) {
  int lane = threadIdx.x;
  int m0 = blockIdx.x << 4, n0 = blockIdx.y << 4, p = blockIdx.z;
  A += (long)p * sA; W += (long)p * sW; bias += (long)p * sB; O += (long)p * sO;
  int mr = lane & 15, hi = lane >> 4;
  const _Float16* ar = A + (long)(m0 + mr) * lda;
  const _Float16* wr = W + (long)(n0 + mr) * K;
  v8f acc = {};
  for (int k0 = 0; k0 < K; k0 += 32) {
    H16 a, b;
    a.h[0] = *(const h8*)(ar + k0 + hi * 8);
    a.h[1] = *(const h8*)(ar + k0 + 16 + hi * 8);
    b.h[0] = *(const h8*)(wr + k0 + hi * 8);
    b.h[1] = *(const h8*)(wr + k0 + 16 + hi * 8);
    acc = wmma_f16(a.v, b.v, acc);
  }
  float bv = bias[n0 + mr];
  #pragma unroll
  for (int r = 0; r < 8; ++r) {
    float v = acc[r] + bv;
    if (relu) v = fmaxf(v, 0.f);
    int m = m0 + r + (hi << 3);
    if (transposed) O[(long)(n0 + mr) * ldo + m] = (_Float16)v;
    else            O[(long)m * ldo + (n0 + mr)] = (_Float16)v;
  }
}

// ---------------- blocked WMMA GEMM (one wave = 32x64 tile, 2x4 WMMA tiles) ----
// mode 0: f16 out + bias; mode 1: f16 out + bias + relu; mode 2: f32 out + bias + residual
__global__ __launch_bounds__(32)
void k_gemm_blk(const _Float16* __restrict__ A, long sA, int lda,
                const _Float16* __restrict__ W, long sW,
                const float* __restrict__ bias, int sB,
                _Float16* __restrict__ O16, float* __restrict__ O32,
                const float* __restrict__ res, long sO, int ldo,
                int K, int mode) {
  int lane = threadIdx.x;
  int m0 = blockIdx.x * 32, n0 = blockIdx.y * 64, p = blockIdx.z;
  A += (long)p * sA; W += (long)p * sW; bias += (long)p * sB;
  int mr = lane & 15, hi = lane >> 4;

  const _Float16* ar[2];
  ar[0] = A + (long)(m0 + mr) * lda;
  ar[1] = ar[0] + 16 * lda;
  const _Float16* wr[4];
  #pragma unroll
  for (int ni = 0; ni < 4; ++ni) wr[ni] = W + (long)(n0 + ni * 16 + mr) * K;

  v8f acc[2][4] = {};
  for (int k0 = 0; k0 < K; k0 += 32) {
    H16 a[2], b[4];
    #pragma unroll
    for (int mi = 0; mi < 2; ++mi) {
      a[mi].h[0] = *(const h8*)(ar[mi] + k0 + hi * 8);
      a[mi].h[1] = *(const h8*)(ar[mi] + k0 + 16 + hi * 8);
    }
    #pragma unroll
    for (int ni = 0; ni < 4; ++ni) {
      b[ni].h[0] = *(const h8*)(wr[ni] + k0 + hi * 8);
      b[ni].h[1] = *(const h8*)(wr[ni] + k0 + 16 + hi * 8);
    }
    #pragma unroll
    for (int mi = 0; mi < 2; ++mi)
      #pragma unroll
      for (int ni = 0; ni < 4; ++ni)
        acc[mi][ni] = wmma_f16(a[mi].v, b[ni].v, acc[mi][ni]);
  }

  #pragma unroll
  for (int ni = 0; ni < 4; ++ni) {
    int n = n0 + ni * 16 + mr;
    float bv = bias[n];
    #pragma unroll
    for (int mi = 0; mi < 2; ++mi) {
      #pragma unroll
      for (int r = 0; r < 8; ++r) {
        int m = m0 + mi * 16 + r + (hi << 3);
        long o = (long)p * sO + (long)m * ldo + n;
        float v = acc[mi][ni][r] + bv;
        if (mode == 2)      O32[o] = v + res[o];
        else if (mode == 1) O16[o] = (_Float16)fmaxf(v, 0.f);
        else                O16[o] = (_Float16)v;
      }
    }
  }
}

// ---------------- fused attention ----------------
// grid (L/64, NH, 4), block 128 (4 waves, one 16-row q tile each)
#define ATT_WAVES 4
__global__ __launch_bounds__(128)
void k_attn(const _Float16* __restrict__ qp, const _Float16* __restrict__ kp,
            const _Float16* __restrict__ vT, _Float16* __restrict__ ao) {
  __shared__ __align__(16) _Float16 smem[ATT_WAVES][16][SP];
  int lane = threadIdx.x & 31, wave = threadIdx.x >> 5;
  int p = blockIdx.z, h = blockIdx.y;
  int l0 = (blockIdx.x * ATT_WAVES + wave) << 4;
  int mr = lane & 15, hi = lane >> 4;
  h8 z8 = {};

  // A fragment: q rows, real K=dh=16 (chunk1 zero)
  const _Float16* qrow = qp + ((long)p * HW + l0 + mr) * CCH + h * 16;
  H16 a; a.h[0] = *(const h8*)(qrow + hi * 8); a.h[1] = z8;

  // scores: 21 tiles of 16 kv tokens
  const _Float16* kbase = kp + (long)p * SS * CCH + h * 16;
  v8f sc[21];
  #pragma unroll
  for (int j = 0; j < 21; ++j) {
    H16 b;
    {
      const _Float16* kr = kbase + (long)(j * 16 + mr) * CCH;
      b.h[0] = *(const h8*)(kr + hi * 8);
      b.h[1] = z8;
    }
    v8f zc = {};
    sc[j] = wmma_f16(a.v, b.v, zc);
  }

  // softmax over s (scale = dh^-0.5 = 0.25)
  #pragma unroll
  for (int r = 0; r < 8; ++r) {
    float mx = -1e30f;
    #pragma unroll
    for (int j = 0; j < 21; ++j) mx = fmaxf(mx, sc[j][r]);
    #pragma unroll
    for (int off = 1; off < 16; off <<= 1) mx = fmaxf(mx, __shfl_xor(mx, off, 32));
    float sum = 0.f;
    #pragma unroll
    for (int j = 0; j < 21; ++j) {
      float e = __expf((sc[j][r] - mx) * 0.25f);
      sc[j][r] = e;
      sum += e;
    }
    #pragma unroll
    for (int off = 1; off < 16; off <<= 1) sum += __shfl_xor(sum, off, 32);
    float inv = 1.f / sum;
    int row = r + (hi << 3);
    #pragma unroll
    for (int j = 0; j < 21; ++j)
      smem[wave][row][j * 16 + mr] = (_Float16)(sc[j][r] * inv);
  }
  // zero-pad columns 336..351
  for (int t = lane; t < 256; t += 32)
    smem[wave][t >> 4][SS + (t & 15)] = (_Float16)0.f;
  __syncthreads();

  // attn @ V : K = 352 (11 steps), N = dh = 16
  v8f acc = {};
  const _Float16* vrow = vT + ((long)p * CCH + h * 16 + mr) * SP;  // column n=mr
  #pragma unroll
  for (int k0 = 0; k0 < SP; k0 += 32) {
    H16 a2, b2;
    a2.h[0] = *(const h8*)(&smem[wave][mr][k0 + hi * 8]);
    a2.h[1] = *(const h8*)(&smem[wave][mr][k0 + 16 + hi * 8]);
    b2.h[0] = *(const h8*)(vrow + k0 + hi * 8);
    b2.h[1] = *(const h8*)(vrow + k0 + 16 + hi * 8);
    acc = wmma_f16(a2.v, b2.v, acc);
  }
  #pragma unroll
  for (int r = 0; r < 8; ++r) {
    int m = l0 + r + (hi << 3);
    ao[((long)p * HW + m) * CCH + h * 16 + mr] = (_Float16)acc[r];
  }
}

// ---------------- LayerNorm (block = 128 threads = one token) ----------------
__global__ void k_ln(const float* __restrict__ y, const float* __restrict__ g,
                     const float* __restrict__ be, float* __restrict__ o32,
                     _Float16* __restrict__ o16, float* __restrict__ img, int mode) {
  __shared__ float red[8];
  int c = threadIdx.x;
  long tok = (long)blockIdx.y * HW + blockIdx.x;
  float v = y[tok * CCH + c];
  float s = v;
  #pragma unroll
  for (int off = 16; off >= 1; off >>= 1) s += __shfl_xor(s, off, 32);
  int wid = c >> 5, lane = c & 31;
  if (!lane) red[wid] = s;
  __syncthreads();
  float mu = (red[0] + red[1] + red[2] + red[3]) * (1.f / 128.f);
  float d = v - mu;
  float s2 = d * d;
  #pragma unroll
  for (int off = 16; off >= 1; off >>= 1) s2 += __shfl_xor(s2, off, 32);
  if (!lane) red[4 + wid] = s2;
  __syncthreads();
  float var = (red[4] + red[5] + red[6] + red[7]) * (1.f / 128.f);
  float o = d * rsqrtf(var + 1e-5f) * g[c] + be[c];
  if (mode == 0) {
    o32[tok * CCH + c] = o;
    o16[tok * CCH + c] = (_Float16)o;
  } else {
    img[((long)blockIdx.y * CCH + c) * HW + blockIdx.x] = o;
  }
}

// glb_out[c,m,n] = x_glb[c,m,n] + src_img[2a+b, c, 2*(m&31), 2*(n&31)]
__global__ void k_glb(const float* __restrict__ x, float* __restrict__ out) {
  int idx = blockIdx.x * 256 + threadIdx.x;   // 128*4096
  int c = idx >> 12;
  int rem = idx & 4095;
  int m = rem >> 6, n = rem & 63;
  int a = m >> 5, b = n >> 5;
  int y = (m & 31) << 1, z = (n & 31) << 1;
  int patch = a * 2 + b;
  long base = (long)4 * CCH * HW;
  out[base + idx] = x[base + idx] + out[((long)patch * CCH + c) * HW + (y << 6) + z];
}

// ---------------- launch ----------------
extern "C" void kernel_launch(void* const* d_in, const int* in_sizes, int n_in,
                              void* d_out, int out_size, void* d_ws, size_t ws_size,
                              hipStream_t stream) {
  (void)in_sizes; (void)n_in; (void)out_size; (void)ws_size;
  const float* x     = (const float*)d_in[0];
  const float* sal_w = (const float*)d_in[1];
  const float* sal_b = (const float*)d_in[2];
  const float* ipw   = (const float*)d_in[3];
  const float* ipb   = (const float*)d_in[4];
  const float* ow    = (const float*)d_in[5];
  const float* ob    = (const float*)d_in[6];
  const float* w3    = (const float*)d_in[7];
  const float* b3    = (const float*)d_in[8];
  const float* w4    = (const float*)d_in[9];
  const float* b4    = (const float*)d_in[10];
  const float* g1    = (const float*)d_in[11];
  const float* be1   = (const float*)d_in[12];
  const float* g2    = (const float*)d_in[13];
  const float* be2   = (const float*)d_in[14];
  float* out = (float*)d_out;

  char* w = (char*)d_ws;
  size_t off = 0;
  auto alloc = [&](size_t bytes) { size_t r = off; off = (off + bytes + 255) & ~(size_t)255; return r; };
  float*    tam   = (float*)   (w + alloc((size_t)HW * 4));
  _Float16* q16   = (_Float16*)(w + alloc((size_t)4 * HW * CCH * 2));
  float*    qf32  = (float*)   (w + alloc((size_t)4 * HW * CCH * 4)); // aliased by hid16 later
  _Float16* Wi16  = (_Float16*)(w + alloc((size_t)4 * 384 * CCH * 2));
  _Float16* Wo16  = (_Float16*)(w + alloc((size_t)4 * CCH * CCH * 2));
  _Float16* w3h   = (_Float16*)(w + alloc((size_t)256 * 128 * 2));
  _Float16* w4h   = (_Float16*)(w + alloc((size_t)128 * 256 * 2));
  _Float16* kv16  = (_Float16*)(w + alloc((size_t)4 * SS * CCH * 2));
  _Float16* qp16  = (_Float16*)(w + alloc((size_t)4 * HW * CCH * 2));
  _Float16* kp16  = (_Float16*)(w + alloc((size_t)4 * SS * CCH * 2));
  _Float16* vT16  = (_Float16*)(w + alloc((size_t)4 * CCH * SP * 2));
  _Float16* ao16  = (_Float16*)(w + alloc((size_t)4 * HW * CCH * 2));
  float*    y1    = (float*)   (w + alloc((size_t)4 * HW * CCH * 4)); // also y2
  float*    src32 = (float*)   (w + alloc((size_t)4 * HW * CCH * 4));
  _Float16* src16 = (_Float16*)(w + alloc((size_t)4 * HW * CCH * 2));
  _Float16* hid16 = (_Float16*)qf32;  // alias: qf32 dead after out-proj

  const float* xg = x + (size_t)4 * CCH * HW;

  // weight conversion
  k_f32_to_f16<<<(4 * 384 * 128 + 255) / 256, 256, 0, stream>>>(ipw, Wi16, 4 * 384 * 128);
  k_f32_to_f16<<<(4 * 128 * 128 + 255) / 256, 256, 0, stream>>>(ow, Wo16, 4 * 128 * 128);
  k_f32_to_f16<<<(256 * 128 + 255) / 256, 256, 0, stream>>>(w3, w3h, 256 * 128);
  k_f32_to_f16<<<(128 * 256 + 255) / 256, 256, 0, stream>>>(w4, w4h, 128 * 256);

  // saliency gate, q tokens, pooled kv
  k_tam<<<HW / 256, 256, 0, stream>>>(xg, sal_w, sal_b, tam);
  k_build_q<<<(4 * CCH * HW) / 256, 256, 0, stream>>>(x, tam, q16, qf32);
  k_pool<<<(4 * SS * CCH + 255) / 256, 256, 0, stream>>>(xg, kv16);
  k_zero16<<<(4 * CCH * SP + 255) / 256, 256, 0, stream>>>(vT16, 4 * CCH * SP);

  // Q projection (blocked 32x64)
  dim3 gq(HW / 32, CCH / 64, 4);
  k_gemm_blk<<<gq, 32, 0, stream>>>(q16, (long)HW * CCH, CCH, Wi16, (long)384 * CCH,
                                    ipb, 384, qp16, nullptr, nullptr,
                                    (long)HW * CCH, CCH, CCH, 0);
  // K / V projections (small)
  dim3 gk(SS / 16, CCH / 16, 4);
  k_gemm_h16<<<gk, 32, 0, stream>>>(kv16, (long)SS * CCH, CCH, Wi16 + 128 * CCH, (long)384 * CCH,
                                    ipb + 128, 384, kp16, (long)SS * CCH, CCH, CCH, 0, 0);
  k_gemm_h16<<<gk, 32, 0, stream>>>(kv16, (long)SS * CCH, CCH, Wi16 + 256 * CCH, (long)384 * CCH,
                                    ipb + 256, 384, vT16, (long)CCH * SP, SP, CCH, 0, 1);

  // fused attention
  dim3 ga(HW / (16 * ATT_WAVES), 8, 4);
  k_attn<<<ga, 32 * ATT_WAVES, 0, stream>>>(qp16, kp16, vT16, ao16);

  // out-proj + bias + residual(q) -> f32
  k_gemm_blk<<<gq, 32, 0, stream>>>(ao16, (long)HW * CCH, CCH, Wo16, (long)CCH * CCH,
                                    ob, CCH, nullptr, y1, qf32,
                                    (long)HW * CCH, CCH, CCH, 2);

  // LN1 -> src (f32 + f16)
  dim3 gl(HW, 4);
  k_ln<<<gl, 128, 0, stream>>>(y1, g1, be1, src32, src16, nullptr, 0);

  // FFN
  dim3 gf1(HW / 32, 256 / 64, 4);
  k_gemm_blk<<<gf1, 32, 0, stream>>>(src16, (long)HW * CCH, CCH, w3h, 0L, b3, 0,
                                     hid16, nullptr, nullptr,
                                     (long)HW * 256, 256, CCH, 1);
  k_gemm_blk<<<gq, 32, 0, stream>>>(hid16, (long)HW * 256, 256, w4h, 0L, b4, 0,
                                    nullptr, y1, src32,
                                    (long)HW * CCH, CCH, 256, 2);

  // LN2 -> d_out (src_img layout), then global branch
  k_ln<<<gl, 128, 0, stream>>>(y1, g2, be2, nullptr, nullptr, out, 1);
  k_glb<<<(CCH * HW) / 256, 256, 0, stream>>>(x, out);
}
